// DetectionModule_36249523978390
// MI455X (gfx1250) — matrix operand order repaired and checked by reading, not compiled, not run on hardware
//
#include <hip/hip_runtime.h>
#include <hip/hip_bf16.h>

// ---------------------------------------------------------------------------
// Detection module, factorized for MI455X (gfx1250, wave32, WMMA).
// GEMMs: bf16 v_wmma_f32_16x16x32_bf16, f32 accumulate, 2x4 register tiling.
// ---------------------------------------------------------------------------

#define BB 32
#define TT 128
#define RR 49
#define RP 64          // R padded to 64 (zero rows)
#define DD 512
#define HH 256
#define THRES 0.1f
#define LAM 4.0f
#define NEG_INF -1e9f

typedef __attribute__((ext_vector_type(16))) __bf16 v16bf;
typedef __attribute__((ext_vector_type(8)))  __bf16 bf16x8;
typedef __attribute__((ext_vector_type(8)))  float  v8f;

// ------------------------------ helpers ------------------------------------

__global__ void f32_to_bf16_kernel(const float* __restrict__ in,
                                   __bf16* __restrict__ out, int n) {
    int i = blockIdx.x * 256 + threadIdx.x;
    if (i < n) out[i] = (__bf16)in[i];
}

// region [B,49,512] f32 -> region_bf [B,64,512] (zero pad) and regionT [B,512,64]
__global__ void prep_region_kernel(const float* __restrict__ reg,
                                   __bf16* __restrict__ reg_bf,
                                   __bf16* __restrict__ regT) {
    int i = blockIdx.x * 256 + threadIdx.x;       // over B*64*512
    int d = i & (DD - 1);
    int r = (i >> 9) & (RP - 1);
    int b = i >> 15;
    float v = (r < RR) ? reg[((long)b * RR + r) * DD + d] : 0.0f;
    reg_bf[i] = (__bf16)v;
    regT[((long)b * DD + d) * RP + r] = (__bf16)v;
}

// in [Dd,Hd] f32 row-major -> out [Hd,Dd] bf16 row-major
__global__ void transpose_to_bf16_kernel(const float* __restrict__ in,
                                         __bf16* __restrict__ out,
                                         int Dd, int Hd) {
    int i = blockIdx.x * 256 + threadIdx.x;
    if (i < Dd * Hd) {
        int d = i % Dd;
        int h = i / Dd;
        out[i] = (__bf16)in[(long)d * Hd + h];
    }
}

// rn [B,64,256] f32 -> rnT [B,256,64] f32 (for coalesced neg-expert reads)
__global__ void transpose_rn_kernel(const float* __restrict__ rn,
                                    float* __restrict__ rnT) {
    int i = blockIdx.x * 256 + threadIdx.x;       // over B*256*64
    int r = i & (RP - 1);
    int h = (i >> 6) & (HH - 1);
    int b = i >> 14;
    rnT[i] = rn[((long)b * RP + r) * HH + h];
}

// ------------------------------ WMMA GEMM ----------------------------------
// C[M,N] = act( A_bf16[M,K] @ (BT_bf16[N,K])^T + bias[N] + addsrc[M,N] )
// One wave computes a 32x64 block (2 M-tiles x 4 N-tiles) of 16x16 WMMA tiles.
// grid = (N/64, M/32, batch). M%32==0, N%64==0, K%32==0.

__global__ __launch_bounds__(32)
void gemm_bf16_wmma_kernel(const __bf16* __restrict__ A,
                           const __bf16* __restrict__ BT,
                           float* __restrict__ C,
                           const float* __restrict__ bias,
                           const float* __restrict__ addsrc,
                           __bf16* __restrict__ outbf,
                           int K, int lda, int ldb, int ldc,
                           long strideA, long strideB, long strideC,
                           int act) {
    int lane = threadIdx.x;            // 0..31
    int tn = blockIdx.x, tm = blockIdx.y, bz = blockIdx.z;
    A  += (long)bz * strideA;
    BT += (long)bz * strideB;
    C  += (long)bz * strideC;

    int half = lane >> 4;              // 0 or 1
    int l16  = lane & 15;

    const __bf16* aptr[2];
    const __bf16* bptr[4];
#pragma unroll
    for (int i = 0; i < 2; ++i)
        aptr[i] = A + (long)(tm * 32 + i * 16 + l16) * lda + half * 8;
#pragma unroll
    for (int j = 0; j < 4; ++j)
        bptr[j] = BT + (long)(tn * 64 + j * 16 + l16) * ldb + half * 8;

    v8f acc[2][4] = {};
    for (int k0 = 0; k0 < K; k0 += 32) {
        // prefetch one k-block ahead (gfx1250 global_prefetch path)
        if (k0 + 32 < K) {
            __builtin_prefetch(aptr[0] + k0 + 32, 0, 1);
            __builtin_prefetch(bptr[0] + k0 + 32, 0, 1);
        }
        v16bf av[2], bv[4];
#pragma unroll
        for (int i = 0; i < 2; ++i) {
            bf16x8 a0 = *(const bf16x8*)(aptr[i] + k0);       // K = k0+half*8 .. +7
            bf16x8 a1 = *(const bf16x8*)(aptr[i] + k0 + 16);  // K = k0+16+half*8 .. +7
#pragma unroll
            for (int e = 0; e < 8; ++e) { av[i][e] = a0[e]; av[i][e + 8] = a1[e]; }
        }
#pragma unroll
        for (int j = 0; j < 4; ++j) {
            bf16x8 b0 = *(const bf16x8*)(bptr[j] + k0);       // K = k0+half*8 .. +7
            bf16x8 b1 = *(const bf16x8*)(bptr[j] + k0 + 16);  // K = k0+16+half*8 .. +7
#pragma unroll
            for (int e = 0; e < 8; ++e) { bv[j][e] = b0[e]; bv[j][e + 8] = b1[e]; }
        }
#pragma unroll
        for (int i = 0; i < 2; ++i)
#pragma unroll
            for (int j = 0; j < 4; ++j)
                acc[i][j] = __builtin_amdgcn_wmma_f32_16x16x32_bf16(
                    false, av[i], false, bv[j], (short)0, acc[i][j], false, false);
    }

#pragma unroll
    for (int i = 0; i < 2; ++i) {
        int crow0 = tm * 32 + i * 16 + half * 8;
#pragma unroll
        for (int j = 0; j < 4; ++j) {
            int ccol = tn * 64 + j * 16 + l16;
            float bval = bias ? bias[ccol] : 0.0f;
#pragma unroll
            for (int v = 0; v < 8; ++v) {
                long idx = (long)(crow0 + v) * ldc + ccol;
                float x = acc[i][j][v] + bval;
                if (addsrc) x += addsrc[(long)bz * strideC + idx];
                if (act == 1) x = tanhf(x);
                C[idx] = x;
                if (outbf) outbf[(long)bz * strideC + idx] = (__bf16)x;
            }
        }
    }
}

// --------------------------- softmax over regions --------------------------
// attn [B,T,64] f32; write attn_pos bf16 [B,T,64] (0 for r>=49) and cnt [B,T].

__global__ void softmax_kernel(const float* __restrict__ attn,
                               __bf16* __restrict__ attn_pos,
                               float* __restrict__ cnt) {
    int t = blockIdx.x, b = blockIdx.y;
    int r = threadIdx.x;               // 0..63
    __shared__ float sa[RP];
    __shared__ int   sneg[RP];
    __shared__ float smax, ssum;
    __shared__ int   scnt;
    long bt = (long)b * TT + t;
    const float* ap = attn + bt * RP;
    float a = (r < RR) ? ap[r] : 0.0f;
    int neg = (r < RR) && ((a - THRES) > 0.0f);
    sa[r] = (neg ? a : NEG_INF) * LAM;
    sneg[r] = neg;
    __syncthreads();
    if (r == 0) {
        float mx = -3.402823466e38f; int c = 0;
        for (int i = 0; i < RR; ++i) { mx = fmaxf(mx, sa[i]); c += sneg[i]; }
        smax = mx; scnt = c;
    }
    __syncthreads();
    float e = (r < RR) ? __expf(sa[r] - smax) : 0.0f;
    sa[r] = e;
    __syncthreads();
    if (r == 0) { float s = 0.0f; for (int i = 0; i < RR; ++i) s += sa[i]; ssum = s; }
    __syncthreads();
    float p = (r < RR) ? e / ssum : 0.0f;
    attn_pos[bt * RP + r] = (__bf16)p;
    if (r == 0) cnt[bt] = (float)scnt;
}

// ------------------------------- pos score ---------------------------------
// pos_score[b,t] = relu( mask * [cnt>0] * sigmoid(ph[b,t,:]@pw2 + pb2) )

__global__ void pos_score_kernel(const float* __restrict__ ph,
                                 const float* __restrict__ pw2,
                                 const float* __restrict__ pb2,
                                 const float* __restrict__ cnt,
                                 const int* __restrict__ amask,
                                 float* __restrict__ pos_score) {
    int t = blockIdx.x, b = blockIdx.y, h = threadIdx.x;
    __shared__ float red[HH];
    long bt = (long)b * TT + t;
    red[h] = ph[bt * HH + h] * pw2[h];
    __syncthreads();
    for (int s = HH / 2; s > 0; s >>= 1) {
        if (h < s) red[h] += red[h + s];
        __syncthreads();
    }
    if (h == 0) {
        float sc = 1.0f / (1.0f + __expf(-(red[0] + pb2[0])));
        if (cnt[bt] == 0.0f) sc = 0.0f;
        sc *= (float)amask[bt];
        pos_score[bt] = fmaxf(sc, 0.0f);
    }
}

// ------------------------------- neg score ---------------------------------
// neg_score[b,t,r] = mask * [attn<=thres] * sigmoid( tanh(wn+rn+nb1)@nw2 + nb2 )
// thread = r; serial reduction over h; rnT coalesced, wn/nb1/nw2 uniform.

__global__ void neg_score_kernel(const float* __restrict__ wn,
                                 const float* __restrict__ rnT,
                                 const float* __restrict__ nb1,
                                 const float* __restrict__ nw2,
                                 const float* __restrict__ nb2,
                                 const float* __restrict__ attn,
                                 const int* __restrict__ amask,
                                 float* __restrict__ neg_score) {
    int t = blockIdx.x, b = blockIdx.y, r = threadIdx.x;   // 64 threads
    long bt = (long)b * TT + t;
    if (r >= RR) {                       // padded region columns
        neg_score[bt * RP + r] = 0.0f;
        return;
    }
    const float* wrow = wn + bt * HH;                 // uniform across lanes
    const float* rcol = rnT + (long)b * HH * RP + r;  // coalesced across lanes
    float acc = 0.0f;
#pragma unroll 8
    for (int h = 0; h < HH; ++h)
        acc += tanhf(wrow[h] + rcol[(long)h * RP] + nb1[h]) * nw2[h];
    float sc = 1.0f / (1.0f + __expf(-(acc + nb2[0])));
    float keep = ((attn[bt * RP + r] - THRES) > 0.0f) ? 0.0f : 1.0f;
    float maskf = (float)amask[bt];
    neg_score[bt * RP + r] = fmaxf(sc * keep * maskf, 0.0f);
}

// row sums of neg_score + total positive count per batch (deterministic)
__global__ void rowsum_kernel(const float* __restrict__ neg_score,
                              const float* __restrict__ cnt,
                              float* __restrict__ row_sum,
                              float* __restrict__ total_cnt) {
    int b = blockIdx.x, t = threadIdx.x;   // 128 threads
    long bt = (long)b * TT + t;
    float s = 0.0f;
    for (int r = 0; r < RR; ++r) s += neg_score[bt * RP + r];
    row_sum[bt] = s;
    __shared__ float red[TT];
    red[t] = cnt[bt];
    __syncthreads();
    for (int k = TT / 2; k > 0; k >>= 1) {
        if (t < k) red[t] += red[t + k];
        __syncthreads();
    }
    if (t == 0) total_cnt[b] = red[0];
}

__global__ void colsum_kernel(const float* __restrict__ neg_score,
                              float* __restrict__ col_sum) {
    int b = blockIdx.x, r = threadIdx.x;   // 64 threads
    float s = 0.0f;
    if (r < RR)
        for (int t = 0; t < TT; ++t) s += neg_score[((long)b * TT + t) * RP + r];
    col_sum[b * RP + r] = s;
}

// pos_agg[b,d] = sum_t pos_score[b,t] * pos[b,t,d]
__global__ void pos_agg_kernel(const float* __restrict__ pos_score,
                               const float* __restrict__ pos,
                               float* __restrict__ pos_agg) {
    int b = blockIdx.y;
    int d = blockIdx.x * 256 + threadIdx.x;
    float acc = 0.0f;
    for (int t = 0; t < TT; ++t)
        acc += pos_score[(long)b * TT + t] * pos[((long)b * TT + t) * DD + d];
    pos_agg[(long)b * DD + d] = acc;
}

// neg_agg[b,d] = sum_t rowsum*word + sum_r colsum*region   (factorized all_emb)
__global__ void neg_agg_kernel(const float* __restrict__ row_sum,
                               const float* __restrict__ col_sum,
                               const float* __restrict__ word,
                               const float* __restrict__ region,
                               float* __restrict__ neg_agg) {
    int b = blockIdx.y;
    int d = blockIdx.x * 256 + threadIdx.x;
    float acc = 0.0f;
    for (int t = 0; t < TT; ++t)
        acc += row_sum[(long)b * TT + t] * word[((long)b * TT + t) * DD + d];
    for (int r = 0; r < RR; ++r)
        acc += col_sum[(long)b * RP + r] * region[((long)b * RR + r) * DD + d];
    neg_agg[(long)b * DD + d] = acc;
}

// ------------------------------- final head --------------------------------
__global__ void final_kernel(const float* __restrict__ neg_agg,
                             const float* __restrict__ pos_agg,
                             const float* __restrict__ total_cnt,
                             const float* __restrict__ fcw,
                             const float* __restrict__ fcb,
                             const float* __restrict__ cw1,
                             const float* __restrict__ cb1,
                             const float* __restrict__ cw2,
                             const float* __restrict__ cb2,
                             float* __restrict__ out) {
    int b = blockIdx.x, h = threadIdx.x;   // 256 threads
    __shared__ float fin[DD];
    __shared__ float red[HH];
    __shared__ float p01[2];
    __shared__ float ch0s, l0s;
    const float* na = neg_agg + (long)b * DD;
    const float* pa = pos_agg + (long)b * DD;

    red[h] = na[h] * fcw[h] + na[h + HH] * fcw[h + HH];
    __syncthreads();
    for (int s = HH / 2; s > 0; s >>= 1) { if (h < s) red[h] += red[h + s]; __syncthreads(); }
    if (h == 0) ch0s = red[0] + fcb[0];
    __syncthreads();
    red[h] = pa[h] * fcw[h] + pa[h + HH] * fcw[h + HH];
    __syncthreads();
    for (int s = HH / 2; s > 0; s >>= 1) { if (h < s) red[h] += red[h + s]; __syncthreads(); }
    if (h == 0) {
        float ch0 = ch0s, ch1 = red[0] + fcb[0];
        if (total_cnt[b] == 0.0f) ch0 = NEG_INF;
        float m  = fmaxf(ch0, ch1);
        float e0 = __expf(ch0 - m), e1 = __expf(ch1 - m);
        float inv = 1.0f / (e0 + e1);
        p01[0] = e0 * inv; p01[1] = e1 * inv;
    }
    __syncthreads();
    float p0 = p01[0], p1 = p01[1];
    fin[h]       = p0 * na[h]       + p1 * pa[h];
    fin[h + HH]  = p0 * na[h + HH]  + p1 * pa[h + HH];
    __syncthreads();

    float acc = cb1[h];
    for (int d = 0; d < DD; ++d) acc += fin[d] * cw1[(long)d * HH + h];
    acc = fmaxf(acc, 0.0f);
    float l0 = acc * cw2[h * 2 + 0];
    float l1 = acc * cw2[h * 2 + 1];
    red[h] = l0;
    __syncthreads();
    for (int s = HH / 2; s > 0; s >>= 1) { if (h < s) red[h] += red[h + s]; __syncthreads(); }
    if (h == 0) l0s = red[0] + cb2[0];
    __syncthreads();
    red[h] = l1;
    __syncthreads();
    for (int s = HH / 2; s > 0; s >>= 1) { if (h < s) red[h] += red[h + s]; __syncthreads(); }
    if (h == 0) {
        out[b * 2 + 0] = l0s;
        out[b * 2 + 1] = red[0] + cb2[1];
        out[BB * 2 + b * 2 + 0] = p01[0];   // path_prob
        out[BB * 2 + b * 2 + 1] = p01[1];
    }
}

// ------------------------------- launcher ----------------------------------

extern "C" void kernel_launch(void* const* d_in, const int* in_sizes, int n_in,
                              void* d_out, int out_size, void* d_ws, size_t ws_size,
                              hipStream_t stream) {
    (void)in_sizes; (void)n_in; (void)out_size; (void)ws_size;
    const float* word   = (const float*)d_in[0];
    const float* region = (const float*)d_in[1];
    const int*   amask  = (const int*)d_in[2];
    const float* pw1 = (const float*)d_in[3];
    const float* pb1 = (const float*)d_in[4];
    const float* pw2 = (const float*)d_in[5];
    const float* pb2 = (const float*)d_in[6];
    const float* nw1 = (const float*)d_in[7];
    const float* nb1 = (const float*)d_in[8];
    const float* nw2 = (const float*)d_in[9];
    const float* nb2 = (const float*)d_in[10];
    const float* fcw = (const float*)d_in[11];
    const float* fcb = (const float*)d_in[12];
    const float* cw1 = (const float*)d_in[13];
    const float* cb1 = (const float*)d_in[14];
    const float* cw2 = (const float*)d_in[15];
    const float* cb2 = (const float*)d_in[16];
    float* out = (float*)d_out;

    char* ws = (char*)d_ws;
    __bf16* word_bf   = (__bf16*)(ws + 0);            // 4,194,304
    __bf16* reg_bf    = (__bf16*)(ws + 4194304);      // 2,097,152
    __bf16* regT_bf   = (__bf16*)(ws + 6291456);      // 2,097,152
    __bf16* nw1T      = (__bf16*)(ws + 8388608);      //   262,144
    __bf16* pw1T      = (__bf16*)(ws + 8650752);      //   262,144
    float*  attn      = (float*)(ws + 8912896);       // 1,048,576
    __bf16* attnpos_bf= (__bf16*)(ws + 9961472);      //   524,288
    float*  pos       = (float*)(ws + 10485760);      // 8,388,608
    __bf16* pos_bf    = (__bf16*)(ws + 18874368);     // 4,194,304
    float*  wn        = (float*)(ws + 23068672);      // 4,194,304
    float*  rn        = (float*)(ws + 27262976);      // 2,097,152
    float*  ph        = (float*)(ws + 29360128);      // 4,194,304
    float*  pos_score = (float*)(ws + 33554432);      //    16,384
    float*  cnt       = (float*)(ws + 33570816);      //    16,384
    float*  total_cnt = (float*)(ws + 33587200);      //       256
    float*  row_sum   = (float*)(ws + 33587456);      //    16,384
    float*  col_sum   = (float*)(ws + 33603840);      //     8,192
    float*  neg_score = (float*)(ws + 33612032);      // 1,048,576
    float*  pos_agg   = (float*)(ws + 34660608);      //    65,536
    float*  neg_agg   = (float*)(ws + 34726144);      //    65,536
    float*  rnT       = (float*)(ws + 34791680);      // 2,097,152

    const int BTD = BB * TT * DD;
    dim3 wblk(32, 1, 1);

    // 1) precision prep
    f32_to_bf16_kernel<<<(BTD + 255) / 256, 256, 0, stream>>>(word, word_bf, BTD);
    prep_region_kernel<<<(BB * RP * DD) / 256, 256, 0, stream>>>(region, reg_bf, regT_bf);
    transpose_to_bf16_kernel<<<(DD * HH) / 256, 256, 0, stream>>>(nw1, nw1T, DD, HH);
    transpose_to_bf16_kernel<<<(DD * HH) / 256, 256, 0, stream>>>(pw1, pw1T, DD, HH);

    // 2) attn[b] = word[b] @ region[b]^T    (M=128, N=64, K=512)
    gemm_bf16_wmma_kernel<<<dim3(RP / 64, TT / 32, BB), wblk, 0, stream>>>(
        word_bf, reg_bf, attn, nullptr, nullptr, nullptr,
        DD, DD, DD, RP, (long)TT * DD, (long)RP * DD, (long)TT * RP, 0);

    // 3) masked softmax over regions
    softmax_kernel<<<dim3(TT, BB), RP, 0, stream>>>(attn, attnpos_bf, cnt);

    // 4) pos[b] = attn_pos[b] @ region[b] + word[b]   (M=128, N=512, K=64)
    gemm_bf16_wmma_kernel<<<dim3(DD / 64, TT / 32, BB), wblk, 0, stream>>>(
        attnpos_bf, regT_bf, pos, nullptr, word, pos_bf,
        RP, RP, RP, DD, (long)TT * RP, (long)DD * RP, (long)TT * DD, 0);

    // 5) wn = word @ nw1 + nb1              (M=4096, N=256, K=512)
    gemm_bf16_wmma_kernel<<<dim3(HH / 64, (BB * TT) / 32, 1), wblk, 0, stream>>>(
        word_bf, nw1T, wn, nb1, nullptr, nullptr,
        DD, DD, DD, HH, 0, 0, 0, 0);

    // 6) rn = region_pad @ nw1 + nb1        (M=2048, N=256, K=512)
    gemm_bf16_wmma_kernel<<<dim3(HH / 64, (BB * RP) / 32, 1), wblk, 0, stream>>>(
        reg_bf, nw1T, rn, nb1, nullptr, nullptr,
        DD, DD, DD, HH, 0, 0, 0, 0);

    // 7) ph = tanh(pos @ pw1 + pb1)         (M=4096, N=256, K=512)
    gemm_bf16_wmma_kernel<<<dim3(HH / 64, (BB * TT) / 32, 1), wblk, 0, stream>>>(
        pos_bf, pw1T, ph, pb1, nullptr, nullptr,
        DD, DD, DD, HH, 0, 0, 0, 1);

    // 7b) transpose rn for coalesced neg-expert reads
    transpose_rn_kernel<<<(BB * HH * RP) / 256, 256, 0, stream>>>(rn, rnT);

    // 8) expert scores
    pos_score_kernel<<<dim3(TT, BB), HH, 0, stream>>>(ph, pw2, pb2, cnt, amask, pos_score);
    neg_score_kernel<<<dim3(TT, BB), RP, 0, stream>>>(wn, rnT, nb1, nw2, nb2, attn, amask, neg_score);

    // 9) deterministic reductions (no atomics)
    rowsum_kernel<<<BB, TT, 0, stream>>>(neg_score, cnt, row_sum, total_cnt);
    colsum_kernel<<<BB, RP, 0, stream>>>(neg_score, col_sum);
    pos_agg_kernel<<<dim3(DD / 256, BB), 256, 0, stream>>>(pos_score, pos, pos_agg);
    neg_agg_kernel<<<dim3(DD / 256, BB), 256, 0, stream>>>(row_sum, col_sum, word, region, neg_agg);

    // 10) path routing + classifier head
    final_kernel<<<BB, HH, 0, stream>>>(neg_agg, pos_agg, total_cnt, fcw, fcb,
                                        cw1, cb1, cw2, cb2, out);
}
